// MTGRN_30210799960411
// MI455X (gfx1250) — compile-verified, matrix operand order checked by reading, not run on hardware
//
#include <hip/hip_runtime.h>
#include <math.h>

// ======================= model dimensions =======================
#define BQ 32
#define TT 12
#define NN 207
#define DD 64
#define NH 8
#define FFD 512
#define MEMN 20
#define RROWS (BQ * TT * NN)   // 79488  (multiple of 128!)
#define BNROWS (BQ * NN)       // 6624
#define CATC 131               // 1 + 2 + 64 + 64
#define GSUP 6
#define ZGC (GSUP * CATC)      // 786

typedef __attribute__((ext_vector_type(16))) _Float16 v16h;
typedef __attribute__((ext_vector_type(8)))  _Float16 v8h;
typedef __attribute__((ext_vector_type(4)))  _Float16 v4h;
typedef __attribute__((ext_vector_type(2)))  _Float16 v2h;
typedef __attribute__((ext_vector_type(8)))  float    v8f;

__device__ __forceinline__ float act_fn(float v, int act) {
  if (act == 1) return fmaxf(v, 0.0f);
  if (act == 2) return 1.0f / (1.0f + __expf(-v));
  if (act == 3) return tanhf(v);
  return v;
}

__device__ __forceinline__ v4h pack4_f16(float a0, float a1, float a2, float a3) {
  // v_cvt_pk_rtz_f16_f32; builtin returns __fp16x2 -> bit_cast to _Float16x2
  v2h p0 = __builtin_bit_cast(v2h, __builtin_amdgcn_cvt_pkrtz(a0, a1));
  v2h p1 = __builtin_bit_cast(v2h, __builtin_amdgcn_cvt_pkrtz(a2, a3));
  v4h h;
  h[0] = p0[0]; h[1] = p0[1]; h[2] = p1[0]; h[3] = p1[1];
  return h;
}

// ======================= WMMA GEMM ==============================
// C[M,N] = act(A[M,K] @ W[K,N] + bias), f32 in/out, f16 WMMA compute,
// f32 accumulate.  Block tile 128x64, K-step 32, 8 waves (4M x 2N);
// each wave owns a 32x32 output tile = four v_wmma_f32_16x16x32_f16
// per K-step.  Staging is branch-free: uniform fast path for interior
// tiles, clamped-address + select path only for edge tiles.
#define GM 128
#define GN 64
#define GK 32
#define LSTR 48   // padded LDS row stride (halves); 96B keeps 16B alignment

__global__ __launch_bounds__(256) void gemm_wmma_f16(
    const float* __restrict__ A, const float* __restrict__ W,
    const float* __restrict__ bias, float* __restrict__ C,
    int M, int N, int K, int lda, int ldw, int ldc,
    int nk, long sAk, long sAb, long sBk, long sBb, long sCk, long sCb,
    int act)
{
  __shared__ __align__(16) _Float16 As[GM * LSTR];   // 12 KB
  __shared__ __align__(16) _Float16 WsT[GN * LSTR];  // 6 KB

  int z  = blockIdx.z;
  int kz = (nk > 1) ? (z % nk) : 0;
  int bz = (nk > 1) ? (z / nk) : z;
  const float* Ab = A + (long)kz * sAk + (long)bz * sAb;
  const float* Wb = W + (long)kz * sBk + (long)bz * sBb;
  float*       Cb = C + (long)kz * sCk + (long)bz * sCb;

  int tid  = threadIdx.x;
  int lane = tid & 31;
  int wid  = tid >> 5;
  int wm   = wid & 3;   // 4 waves in M: 32 rows each
  int wn   = wid >> 2;  // 2 waves in N: 32 cols each
  int m0   = blockIdx.x * GM;
  int n0   = blockIdx.y * GN;

  const bool fullM = (m0 + GM) <= M;
  const bool fullN = (n0 + GN) <= N;

  v8f acc00 = {}, acc01 = {}, acc10 = {}, acc11 = {};

  for (int k0 = 0; k0 < K; k0 += GK) {
    const bool fullK = (k0 + GK) <= K;

    // ---- stage A tile 128x32 (f32 -> f16), 4 contiguous elems/thread ----
    if (fullM && fullK) {
      #pragma unroll
      for (int i = 0; i < (GM * GK) / (256 * 4); ++i) {   // 4
        int g4 = i * 256 + tid;
        int r  = g4 >> 3;            // 8 groups of 4 per 32-wide row
        int c  = (g4 & 7) * 4;
        const float* rp = Ab + (long)(m0 + r) * lda + k0 + c;
        *(v4h*)&As[r * LSTR + c] = pack4_f16(rp[0], rp[1], rp[2], rp[3]);
      }
    } else {
      #pragma unroll
      for (int i = 0; i < (GM * GK) / (256 * 4); ++i) {
        int g4 = i * 256 + tid;
        int r  = g4 >> 3;
        int c  = (g4 & 7) * 4;
        int gm = m0 + r;
        bool rok = gm < M;
        const float* rp = Ab + (long)(rok ? gm : 0) * lda;
        bool okf[4];
        int  kcl[4];
        float vv[4];
        #pragma unroll
        for (int j = 0; j < 4; ++j) {        // clamped addresses / flags
          int kk = k0 + c + j;
          okf[j] = rok && (kk < K);
          kcl[j] = (kk < K) ? kk : 0;
        }
        #pragma unroll
        for (int j = 0; j < 4; ++j) vv[j] = rp[kcl[j]];   // 4 indep loads
        #pragma unroll
        for (int j = 0; j < 4; ++j) vv[j] = okf[j] ? vv[j] : 0.0f;
        *(v4h*)&As[r * LSTR + c] = pack4_f16(vv[0], vv[1], vv[2], vv[3]);
      }
    }

    // ---- stage W tile 32x64 transposed -> WsT[n][k] ----
    if (fullN && fullK) {
      #pragma unroll
      for (int i = 0; i < (GN * GK) / (256 * 4); ++i) {   // 2
        int g4 = i * 256 + tid;
        int r  = g4 >> 4;            // K row (16 groups of 4 per 64-wide row)
        int c  = (g4 & 15) * 4;      // N col
        const float* rp = Wb + (long)(k0 + r) * ldw + n0 + c;
        WsT[(c + 0) * LSTR + r] = (_Float16)rp[0];
        WsT[(c + 1) * LSTR + r] = (_Float16)rp[1];
        WsT[(c + 2) * LSTR + r] = (_Float16)rp[2];
        WsT[(c + 3) * LSTR + r] = (_Float16)rp[3];
      }
    } else {
      #pragma unroll
      for (int i = 0; i < (GN * GK) / (256 * 4); ++i) {
        int g4 = i * 256 + tid;
        int r  = g4 >> 4;
        int c  = (g4 & 15) * 4;
        int gk = k0 + r;
        bool kok = gk < K;
        const float* rp = Wb + (long)(kok ? gk : 0) * ldw;
        bool okf[4];
        int  ncl[4];
        float vv[4];
        #pragma unroll
        for (int j = 0; j < 4; ++j) {
          int gn = n0 + c + j;
          okf[j] = kok && (gn < N);
          ncl[j] = (gn < N) ? gn : 0;
        }
        #pragma unroll
        for (int j = 0; j < 4; ++j) vv[j] = rp[ncl[j]];
        #pragma unroll
        for (int j = 0; j < 4; ++j)
          WsT[(c + j) * LSTR + r] = (_Float16)(okf[j] ? vv[j] : 0.0f);
      }
    }
    __syncthreads();

    // CDNA5 prefetch of next A tile (global_prefetch_b8)
    if (k0 + GK < K) {
      int pr = m0 + (tid >> 1);
      if (pr < M) __builtin_prefetch(&Ab[(long)pr * lda + k0 + GK], 0, 1);
    }

    // ---- build WMMA fragments per ISA 7.12.2 16-bit layout ----
    // A 16x32: lanes 0-15 (row=lane) carry K {0..7,16..23}; lanes 16-31
    // (row=lane-16) carry K {8..15,24..31}.  B mirrors with N instead of M.
    int hs   = (lane >> 4) << 3;            // 0 or 8 halves
    int arow = wm * 32 + (lane & 15);
    v8h a0lo = *(const v8h*)&As[arow * LSTR + hs];
    v8h a0hi = *(const v8h*)&As[arow * LSTR + 16 + hs];
    v8h a1lo = *(const v8h*)&As[(arow + 16) * LSTR + hs];
    v8h a1hi = *(const v8h*)&As[(arow + 16) * LSTR + 16 + hs];
    int bcol = wn * 32 + (lane & 15);
    v8h b0lo = *(const v8h*)&WsT[bcol * LSTR + hs];
    v8h b0hi = *(const v8h*)&WsT[bcol * LSTR + 16 + hs];
    v8h b1lo = *(const v8h*)&WsT[(bcol + 16) * LSTR + hs];
    v8h b1hi = *(const v8h*)&WsT[(bcol + 16) * LSTR + 16 + hs];
    v16h av0, av1, bv0, bv1;
    #pragma unroll
    for (int i = 0; i < 8; ++i) {
      av0[i] = a0lo[i]; av0[8 + i] = a0hi[i];
      av1[i] = a1lo[i]; av1[8 + i] = a1hi[i];
      bv0[i] = b0lo[i]; bv0[8 + i] = b0hi[i];
      bv1[i] = b1lo[i]; bv1[8 + i] = b1hi[i];
    }
    acc00 = __builtin_amdgcn_wmma_f32_16x16x32_f16(false, av0, false, bv0,
                                                   (short)0, acc00, false, false);
    acc01 = __builtin_amdgcn_wmma_f32_16x16x32_f16(false, av0, false, bv1,
                                                   (short)0, acc01, false, false);
    acc10 = __builtin_amdgcn_wmma_f32_16x16x32_f16(false, av1, false, bv0,
                                                   (short)0, acc10, false, false);
    acc11 = __builtin_amdgcn_wmma_f32_16x16x32_f16(false, av1, false, bv1,
                                                   (short)0, acc11, false, false);
    __syncthreads();
  }

  // ---- epilogue: C/D layout = VGPR r, lane<16 -> M=r, lane>=16 -> M=r+8 ----
  int cn0 = n0 + wn * 32 + (lane & 15);
  #pragma unroll
  for (int sn = 0; sn < 2; ++sn) {
    int cn = cn0 + sn * 16;
    if (cn < N) {
      float bval = bias ? bias[cn] : 0.0f;
      #pragma unroll
      for (int r = 0; r < 8; ++r) {
        int mrow = m0 + wm * 32 + ((lane < 16) ? r : r + 8);
        float v0 = (sn == 0) ? acc00[r] : acc01[r];
        float v1 = (sn == 0) ? acc10[r] : acc11[r];
        if (mrow < M)
          Cb[(long)mrow * ldc + cn] = act_fn(v0 + bval, act);
        if (mrow + 16 < M)
          Cb[(long)(mrow + 16) * ldc + cn] = act_fn(v1 + bval, act);
      }
    }
  }
}

// ======================= embedding ==============================
__global__ void embed_kernel(const float* __restrict__ x, const float* __restrict__ xcov,
                             const float* __restrict__ vlW, const float* __restrict__ vlb,
                             const float* __restrict__ day, const float* __restrict__ week,
                             const float* __restrict__ adp, float* __restrict__ h)
{
  long r = (long)blockIdx.x * blockDim.x + threadIdx.x;
  if (r >= (long)RROWS) return;
  int n = (int)(r % NN);
  long t2 = r / NN;
  int t = (int)(t2 % TT);
  float xv = x[r];
  float minute = xcov[2 * r + 0];
  int wi = (int)xcov[2 * r + 1];
  int di = (int)rintf(minute * 1440.0f);
  di = di < 0 ? 0 : (di > 1439 ? 1439 : di);
  float* hr = h + r * 64;
  const float kln = 9.210340371976184f; // ln(10000)
  #pragma unroll
  for (int c = 0; c < 12; ++c) {
    int j = c >> 1;
    float f = __expf(-(float)(2 * j) * (kln / 12.0f));
    float pe = (c & 1) ? __cosf((float)t * f) : __sinf((float)t * f);
    hr[c] = xv * vlW[c] + vlb[c] + pe;
  }
  #pragma unroll
  for (int c = 0; c < 12; ++c) hr[12 + c] = day[di * 12 + c];
  #pragma unroll
  for (int c = 0; c < 12; ++c) hr[24 + c] = week[wi * 12 + c];
  #pragma unroll
  for (int c = 0; c < 28; ++c) hr[36 + c] = adp[((long)t * NN + n) * 28 + c];
}

// ===== transpose [B,d1,d2,64] -> [B,d2,d1,64] (one 64-ch row/block) =====
__global__ void swap12_kernel(const float* __restrict__ in, float* __restrict__ out,
                              int d1, int d2)
{
  long r = blockIdx.x;
  int c = threadIdx.x;
  long n2 = r % d2;
  long tmp = r / d2;
  long n1 = tmp % d1;
  long b = tmp / d1;
  out[((((b * d2 + n2) * d1) + n1) << 6) + c] = in[(r << 6) + c];
}

// ======================= attention core =========================
// one block per (sequence, head): K/V head slices staged in LDS via
// CDNA5 async global->LDS copies (ASYNCcnt), each thread owns one
// query row with online softmax (hd = 8).
__global__ __launch_bounds__(256) void attn_core_kernel(
    const float* __restrict__ q, const float* __restrict__ k,
    const float* __restrict__ v, float* __restrict__ o, int L)
{
  int s = blockIdx.x;
  int h = blockIdx.y;
  __shared__ float kh[NN * 8];
  __shared__ float vh[NN * 8];
  long base = (long)s * L * 64 + h * 8;
  // async global->LDS staging: per-lane LDS byte offset (low 32 bits of
  // the generic LDS address) + 64-bit global address pair.
  for (int idx = threadIdx.x; idx < L * 8; idx += 256) {
    int row = idx >> 3, c = idx & 7;
    unsigned lk = (unsigned)(unsigned long long)&kh[idx];
    unsigned lv = (unsigned)(unsigned long long)&vh[idx];
    const float* gk = k + base + (long)row * 64 + c;
    const float* gv = v + base + (long)row * 64 + c;
    asm volatile("global_load_async_to_lds_b32 %0, %1, off"
                 :: "v"(lk), "v"(gk) : "memory");
    asm volatile("global_load_async_to_lds_b32 %0, %1, off"
                 :: "v"(lv), "v"(gv) : "memory");
  }
  asm volatile("s_wait_asynccnt 0" ::: "memory");
  __syncthreads();
  int qi = threadIdx.x;
  if (qi >= L) return;
  float qr[8];
  #pragma unroll
  for (int c = 0; c < 8; ++c) qr[c] = q[base + (long)qi * 64 + c];
  float mx = -1e30f, l = 0.0f;
  float acc[8] = {0, 0, 0, 0, 0, 0, 0, 0};
  const float scale = 0.35355339059327373f; // 1/sqrt(8)
  for (int j = 0; j < L; ++j) {
    float sd = 0.0f;
    #pragma unroll
    for (int c = 0; c < 8; ++c) sd += qr[c] * kh[j * 8 + c];
    sd *= scale;
    float mn = fmaxf(mx, sd);
    float sc = __expf(mx - mn);
    float p = __expf(sd - mn);
    l = l * sc + p;
    #pragma unroll
    for (int c = 0; c < 8; ++c) acc[c] = acc[c] * sc + p * vh[j * 8 + c];
    mx = mn;
  }
  float inv = 1.0f / l;
  #pragma unroll
  for (int c = 0; c < 8; ++c) o[base + (long)qi * 64 + c] = acc[c] * inv;
}

// ======================= residual + LN ==========================
__global__ void add_ln_kernel(const float* __restrict__ xa, const float* __restrict__ xb,
                              const float* __restrict__ g, const float* __restrict__ be,
                              float* __restrict__ out, int R)
{
  int r = blockIdx.x * blockDim.x + threadIdx.x;
  if (r >= R) return;
  long o = (long)r * 64;
  float buf[64];
  float s = 0.0f;
  #pragma unroll
  for (int i = 0; i < 64; ++i) { float v = xa[o + i] + xb[o + i]; buf[i] = v; s += v; }
  float mean = s * (1.0f / 64.0f);
  float var = 0.0f;
  #pragma unroll
  for (int i = 0; i < 64; ++i) { float d = buf[i] - mean; var += d * d; }
  var *= (1.0f / 64.0f);
  float inv = rsqrtf(var + 1e-5f);
  #pragma unroll
  for (int i = 0; i < 64; ++i) out[o + i] = (buf[i] - mean) * inv * g[i] + be[i];
}

// ======================= memory query ===========================
// value = softmax(q@Mem^T)@Mem with online softmax, plus top-3 gather.
// Input rows indexed (b,d1,d2); outputs optionally swap (d1,d2).
__global__ __launch_bounds__(128) void memquery_kernel(
    const float* __restrict__ qin, const float* __restrict__ Mem, int Mrows,
    float* __restrict__ value, float* __restrict__ qout,
    float* __restrict__ pos, float* __restrict__ n1o, float* __restrict__ n2o,
    int d1, int d2, int doSwap)
{
  long r = (long)blockIdx.x * blockDim.x + threadIdx.x;
  if (r >= (long)BQ * d1 * d2) return;
  long qoff = r << 6;
  float qr[64];
  #pragma unroll
  for (int i = 0; i < 64; ++i) qr[i] = qin[qoff + i];
  float mx = -1e30f, l = 0.0f;
  float acc[64];
  #pragma unroll
  for (int i = 0; i < 64; ++i) acc[i] = 0.0f;
  float t0 = -1e30f, t1 = -1e30f, t2 = -1e30f;
  int i0 = 0, i1 = 0, i2 = 0;
  for (int m = 0; m < Mrows; ++m) {
    const float* mr = Mem + (long)m * 64;
    float s = 0.0f;
    #pragma unroll
    for (int i = 0; i < 64; ++i) s += qr[i] * mr[i];
    if (s > t0)      { t2 = t1; i2 = i1; t1 = t0; i1 = i0; t0 = s; i0 = m; }
    else if (s > t1) { t2 = t1; i2 = i1; t1 = s; i1 = m; }
    else if (s > t2) { t2 = s; i2 = m; }
    float mn = fmaxf(mx, s);
    float sc = __expf(mx - mn);
    float p = __expf(s - mn);
    l = l * sc + p;
    #pragma unroll
    for (int i = 0; i < 64; ++i) acc[i] = acc[i] * sc + p * mr[i];
    mx = mn;
  }
  float inv = 1.0f / l;
  #pragma unroll
  for (int i = 0; i < 64; ++i) value[qoff + i] = acc[i] * inv;  // input layout
  long r2 = r;
  if (doSwap) {
    long c2 = r % d2;
    long tt = r / d2;
    long c1 = tt % d1;
    long b = tt / d1;
    r2 = (b * d2 + c2) * d1 + c1;
  }
  long oo = r2 << 6;
  const float* m0p = Mem + (long)i0 * 64;
  const float* m1p = Mem + (long)i1 * 64;
  const float* m2p = Mem + (long)i2 * 64;
  #pragma unroll
  for (int i = 0; i < 64; ++i) {
    qout[oo + i] = qr[i];
    pos[oo + i] = m0p[i];
    n1o[oo + i] = m1p[i];
    n2o[oo + i] = m2p[i];
  }
}

// ======== fuse: x_fusion = x_s + x_t ; h0 = sum_t 2*x_t*w[t]+b ========
__global__ void fuse_ht_kernel(const float* __restrict__ vs /*[B,T,N,64]*/,
                               const float* __restrict__ vt /*[B,N,T,64]*/,
                               const float* __restrict__ regw, const float* __restrict__ regb,
                               float* __restrict__ xf /*[B,T,N,64]*/, float* __restrict__ hstate)
{
  long idx = (long)blockIdx.x * blockDim.x + threadIdx.x;
  if (idx >= (long)BNROWS * 64) return;
  int c = (int)(idx & 63);
  long rr = idx >> 6;
  int n = (int)(rr % NN);
  int b = (int)(rr / NN);
  float s = 0.0f;
  #pragma unroll
  for (int t = 0; t < TT; ++t) {
    float vtv = vt[(((long)b * NN + n) * TT + t) * 64 + c];
    s += 2.0f * vtv * regw[t];
    long o = (((long)b * TT + t) * NN + n) * 64 + c;
    xf[o] = vs[o] + vtv;
  }
  hstate[idx] = s + regb[0];
}

// ======================= graph supports =========================
__global__ void calc_e_kernel(const float* __restrict__ We, const float* __restrict__ Mem,
                              float* __restrict__ e)
{
  int idx = blockIdx.x * blockDim.x + threadIdx.x;
  if (idx >= NN * 64) return;
  int n = idx >> 6, c = idx & 63;
  float s = 0.0f;
  #pragma unroll
  for (int m = 0; m < MEMN; ++m) s += We[n * MEMN + m] * Mem[m * 64 + c];
  e[idx] = s;
}

__global__ __launch_bounds__(256) void calc_g_kernel(const float* __restrict__ ea,
                                                     const float* __restrict__ eb,
                                                     float* __restrict__ g)
{
  int n = blockIdx.x;
  __shared__ float er[64];
  __shared__ float red[256];
  for (int i = threadIdx.x; i < 64; i += 256) er[i] = ea[n * 64 + i];
  __syncthreads();
  int j = threadIdx.x;
  float val = 0.0f;
  float mval = -1e30f;
  if (j < NN) {
    float d = 0.0f;
    #pragma unroll
    for (int c = 0; c < 64; ++c) d += er[c] * eb[j * 64 + c];
    val = fmaxf(d, 0.0f);
    mval = val;
  }
  red[threadIdx.x] = mval;
  __syncthreads();
  for (int st = 128; st > 0; st >>= 1) {
    if (threadIdx.x < st) red[threadIdx.x] = fmaxf(red[threadIdx.x], red[threadIdx.x + st]);
    __syncthreads();
  }
  float mx = red[0];
  __syncthreads();
  float e = (j < NN) ? __expf(val - mx) : 0.0f;
  red[threadIdx.x] = e;
  __syncthreads();
  for (int st = 128; st > 0; st >>= 1) {
    if (threadIdx.x < st) red[threadIdx.x] += red[threadIdx.x + st];
    __syncthreads();
  }
  float inv = 1.0f / red[0];
  if (j < NN) g[n * NN + j] = e * inv;
}

__global__ void build_G_kernel(const float* __restrict__ g1, const float* __restrict__ g2,
                               float* __restrict__ G)
{
  int idx = blockIdx.x * blockDim.x + threadIdx.x;
  if (idx >= NN * NN) return;
  int n = idx / NN, m = idx % NN;
  float I = (n == m) ? 1.0f : 0.0f;
  float a = 0.0f, b2 = 0.0f;
  for (int k = 0; k < NN; ++k) {
    a += g1[n * NN + k] * g1[k * NN + m];
    b2 += g2[n * NN + k] * g2[k * NN + m];
  }
  G[0 * NN * NN + idx] = I;
  G[1 * NN * NN + idx] = g1[idx];
  G[2 * NN * NN + idx] = 2.0f * a - I;
  G[3 * NN * NN + idx] = I;
  G[4 * NN * NN + idx] = g2[idx];
  G[5 * NN * NN + idx] = 2.0f * b2 - I;
}

// ======================= decoder helpers ========================
__global__ void zero_kernel(float* __restrict__ p, long n) {
  long i = (long)blockIdx.x * blockDim.x + threadIdx.x;
  if (i < n) p[i] = 0.0f;
}

__global__ void dec_cat_kernel(const float* __restrict__ go, const float* __restrict__ ycov,
                               const float* __restrict__ xf, const float* __restrict__ hstate,
                               const float* __restrict__ zr, float* __restrict__ cat,
                               int t, int useZ)
{
  long idx = (long)blockIdx.x * blockDim.x + threadIdx.x;
  if (idx >= (long)BNROWS * CATC) return;
  int c = (int)(idx % CATC);
  long rr = idx / CATC;
  int n = (int)(rr % NN);
  int b = (int)(rr / NN);
  long btn = ((long)b * TT + t) * NN + n;
  float v;
  if (c == 0) v = go[rr];
  else if (c < 3) v = ycov[btn * 2 + (c - 1)];
  else if (c < 67) v = xf[btn * 64 + (c - 3)];
  else {
    float h = hstate[rr * 64 + (c - 67)];
    v = useZ ? zr[rr * 128 + (c - 67)] * h : h;
  }
  cat[idx] = v;
}

__global__ void dec_update_kernel(const float* __restrict__ zr, const float* __restrict__ hc,
                                  float* __restrict__ hstate, const float* __restrict__ projW,
                                  const float* __restrict__ projb, float* __restrict__ go,
                                  float* __restrict__ out0, int t)
{
  long r = (long)blockIdx.x * blockDim.x + threadIdx.x;
  if (r >= (long)BNROWS) return;
  float acc = 0.0f;
  #pragma unroll
  for (int i = 0; i < 64; ++i) {
    float rg = zr[r * 128 + 64 + i];
    float h = hstate[r * 64 + i];
    float hn = rg * h + (1.0f - rg) * hc[r * 64 + i];
    hstate[r * 64 + i] = hn;
    acc += hn * projW[i];
  }
  float g = acc + projb[0];
  go[r] = g;
  int n = (int)(r % NN);
  int b = (int)(r / NN);
  out0[((long)b * TT + t) * NN + n] = g;
}

// ======================= host side ==============================
static void launch_gemm(hipStream_t stream,
    const float* A, const float* W, const float* bias, float* C,
    int M, int N, int K, int lda, int ldw, int ldc, int act,
    int gz = 1, int nk = 1,
    long sAk = 0, long sAb = 0, long sBk = 0, long sBb = 0,
    long sCk = 0, long sCb = 0)
{
  dim3 grid((M + GM - 1) / GM, (N + GN - 1) / GN, gz);
  gemm_wmma_f16<<<grid, dim3(256), 0, stream>>>(
      A, W, bias, C, M, N, K, lda, ldw, ldc, nk, sAk, sAb, sBk, sBb, sCk, sCb, act);
}

// one post-LN transformer block on [S, L, 64] rows (R = S*L)
static void run_attn_layer(hipStream_t stream, const float* const* P,
                           float* xbuf, float* q, float* k, float* v,
                           float* o, float* ffmid, int S, int L)
{
  int R = S * L;
  launch_gemm(stream, xbuf, P[0], P[1], q, R, 64, 64, 64, 64, 64, 0);
  launch_gemm(stream, xbuf, P[2], P[3], k, R, 64, 64, 64, 64, 64, 0);
  launch_gemm(stream, xbuf, P[4], P[5], v, R, 64, 64, 64, 64, 64, 0);
  attn_core_kernel<<<dim3(S, NH), 256, 0, stream>>>(q, k, v, o, L);
  launch_gemm(stream, o, P[6], P[7], q, R, 64, 64, 64, 64, 64, 0);
  add_ln_kernel<<<(R + 255) / 256, 256, 0, stream>>>(xbuf, q, P[12], P[13], xbuf, R);
  launch_gemm(stream, xbuf, P[8], P[9], ffmid, R, FFD, 64, 64, FFD, FFD, 1);
  launch_gemm(stream, ffmid, P[10], P[11], q, R, 64, FFD, FFD, 64, 64, 0);
  add_ln_kernel<<<(R + 255) / 256, 256, 0, stream>>>(xbuf, q, P[14], P[15], xbuf, R);
}

extern "C" void kernel_launch(void* const* d_in, const int* in_sizes, int n_in,
                              void* d_out, int out_size, void* d_ws, size_t ws_size,
                              hipStream_t stream) {
  (void)in_sizes; (void)n_in; (void)out_size;

  // ---- inputs (setup_inputs dict insertion order, nested dicts in order) ----
  const float* x    = (const float*)d_in[0];
  const float* xcov = (const float*)d_in[1];
  const float* ycov = (const float*)d_in[2];
  const float* vlW  = (const float*)d_in[3];
  const float* vlb  = (const float*)d_in[4];
  const float* day  = (const float*)d_in[5];
  const float* week = (const float*)d_in[6];
  const float* adp  = (const float*)d_in[7];
  const float* TFp[16]; for (int i = 0; i < 16; ++i) TFp[i] = (const float*)d_in[8 + i];
  const float* SFp[16]; for (int i = 0; i < 16; ++i) SFp[i] = (const float*)d_in[24 + i];
  const float* memT_Mem = (const float*)d_in[40];
  const float* memT_Wq  = (const float*)d_in[41];
  const float* We1      = (const float*)d_in[42];
  const float* We2      = (const float*)d_in[43];
  const float* memS_Mem = (const float*)d_in[44];
  const float* memS_Wq  = (const float*)d_in[45];
  const float* regw     = (const float*)d_in[46];
  const float* regb     = (const float*)d_in[47];
  const float* Wg       = (const float*)d_in[48];
  const float* bg       = (const float*)d_in[49];
  const float* Wu       = (const float*)d_in[50];
  const float* bu       = (const float*)d_in[51];
  const float* projW    = (const float*)d_in[52];
  const float* projb    = (const float*)d_in[53];

  // ---- workspace layout (floats) ----
  const size_t RB = (size_t)RROWS * 64;          // 5,087,232
  const size_t need = 6 * RB + (size_t)RROWS * FFD;
  if (ws_size < need * sizeof(float)) return;
  float* buf0 = (float*)d_ws;        // h / x_sr        [B,T,N,64]
  float* buf1 = buf0 + RB;           // ht / x_tr       [B,N,T,64]
  float* buf2 = buf1 + RB;           // q / tmp
  float* buf3 = buf2 + RB;           // k / value_s     [B,T,N,64]
  float* buf4 = buf3 + RB;           // v / value_t     [B,N,T,64]
  float* buf5 = buf4 + RB;           // attn o / x_fusion
  float* buf6 = buf5 + RB;           // FF mid [R,512]; reused by decoder area
  // decoder / graph area overlaid on buf6 (used only after FF is done)
  float* e1  = buf6;
  float* e2  = e1 + NN * 64;
  float* g1  = e2 + NN * 64;
  float* g2  = g1 + NN * NN;
  float* Gb  = g2 + NN * NN;                        // [6,207,207]
  float* hst = Gb + GSUP * NN * NN;                 // [6624,64]
  float* goB = hst + (size_t)BNROWS * 64;           // [6624]
  float* cat = goB + BNROWS;                        // [6624,131]
  float* ZG  = cat + (size_t)BNROWS * CATC;         // [6624,786]
  float* zrB = ZG + (size_t)BNROWS * ZGC;           // [6624,128]
  float* hcB = zrB + (size_t)BNROWS * 128;          // [6624,64]

  // ---- output layout (tuple, flat concat) ----
  const size_t P = RB;  // 79488*64
  float* out = (float*)d_out;
  float* out0  = out;
  float* oqs   = out + RROWS;
  float* opos  = oqs + P;
  float* ons1  = opos + P;
  float* ons2  = ons1 + P;
  float* oqt   = ons2 + P;
  float* opost = oqt + P;
  float* ont1  = opost + P;
  float* ont2  = ont1 + P;

  // ---- 1. embedding -> h [B,T,N,64] ----
  embed_kernel<<<(RROWS + 255) / 256, 256, 0, stream>>>(x, xcov, vlW, vlb, day, week, adp, buf0);

  // ---- 2. temporal transformer on ht [B,N,T,64] ----
  swap12_kernel<<<RROWS, 64, 0, stream>>>(buf0, buf1, TT, NN);
  run_attn_layer(stream, TFp, buf1, buf2, buf3, buf4, buf5, buf6, BQ * NN, TT);

  // ---- 3. spatial transformer on h [B,T,N,64] ----
  run_attn_layer(stream, SFp, buf0, buf2, buf3, buf4, buf5, buf6, BQ * TT, NN);

  // ---- 4. spatial memory query (no layout swap) ----
  launch_gemm(stream, buf0, memS_Wq, nullptr, buf2, RROWS, 64, 64, 64, 64, 64, 0);
  memquery_kernel<<<(RROWS + 127) / 128, 128, 0, stream>>>(
      buf2, memS_Mem, NN, buf3, oqs, opos, ons1, ons2, TT, NN, 0);

  // ---- 5. temporal memory query (swap [B,N,T]->[B,T,N] for outputs) ----
  launch_gemm(stream, buf1, memT_Wq, nullptr, buf2, RROWS, 64, 64, 64, 64, 64, 0);
  memquery_kernel<<<(RROWS + 127) / 128, 128, 0, stream>>>(
      buf2, memT_Mem, MEMN, buf4, oqt, opost, ont1, ont2, NN, TT, 1);

  // ---- 6. fusion + initial decoder state ----
  fuse_ht_kernel<<<((long)BNROWS * 64 + 255) / 256, 256, 0, stream>>>(
      buf3, buf4, regw, regb, buf5, hst);

  // ---- 7. adaptive graph supports ----
  calc_e_kernel<<<(NN * 64 + 255) / 256, 256, 0, stream>>>(We1, memT_Mem, e1);
  calc_e_kernel<<<(NN * 64 + 255) / 256, 256, 0, stream>>>(We2, memT_Mem, e2);
  calc_g_kernel<<<NN, 256, 0, stream>>>(e1, e2, g1);
  calc_g_kernel<<<NN, 256, 0, stream>>>(e2, e1, g2);
  build_G_kernel<<<(NN * NN + 255) / 256, 256, 0, stream>>>(g1, g2, Gb);

  // ---- 8. GRU decoder, 12 sequential steps ----
  zero_kernel<<<(BNROWS + 255) / 256, 256, 0, stream>>>(goB, BNROWS);
  const long catElems = (long)BNROWS * CATC;
  for (int t = 0; t < TT; ++t) {
    // gates: zr = sigmoid( gconv(cat(xt,h)) @ Wg + bg )
    dec_cat_kernel<<<(catElems + 255) / 256, 256, 0, stream>>>(
        goB, ycov, buf5, hst, zrB, cat, t, 0);
    launch_gemm(stream, Gb, cat, nullptr, ZG, NN, CATC, NN, NN, CATC, ZGC, 0,
                GSUP * BQ, GSUP,
                (long)NN * NN, 0, 0, (long)NN * CATC, (long)CATC, (long)NN * ZGC);
    launch_gemm(stream, ZG, Wg, bg, zrB, BNROWS, 128, ZGC, ZGC, 128, 128, 2);
    // candidate: hc = tanh( gconv(cat(xt, z*h)) @ Wu + bu )
    dec_cat_kernel<<<(catElems + 255) / 256, 256, 0, stream>>>(
        goB, ycov, buf5, hst, zrB, cat, t, 1);
    launch_gemm(stream, Gb, cat, nullptr, ZG, NN, CATC, NN, NN, CATC, ZGC, 0,
                GSUP * BQ, GSUP,
                (long)NN * NN, 0, 0, (long)NN * CATC, (long)CATC, (long)NN * ZGC);
    launch_gemm(stream, ZG, Wu, bu, hcB, BNROWS, 64, ZGC, ZGC, 64, 64, 3);
    // state update + projection
    dec_update_kernel<<<(BNROWS + 255) / 256, 256, 0, stream>>>(
        zrB, hcB, hst, projW, projb, goB, out0, t);
  }
}
